// GANetHeadFast_53042846105740
// MI455X (gfx1250) — matrix operand order repaired and checked by reading, not compiled, not run on hardware
//
#include <hip/hip_runtime.h>
#include <hip/hip_bf16.h>

#define Hh 160
#define Ww 400
#define Cc 256
#define HW (Hh*Ww)          // 64000
#define WP (Ww+2)           // 402 padded width
#define HP (Hh+2)           // 162 padded height
#define KTOT (9*Cc)         // 2304

typedef __attribute__((ext_vector_type(16))) __bf16 v16bf;
typedef __attribute__((ext_vector_type(8)))  float  v8f;

union FragU { uint4 u[2]; v16bf v; };

__device__ __forceinline__ unsigned short f2bf(float f) {
    unsigned int u = __float_as_uint(f);
    u += 0x7fffu + ((u >> 16) & 1u);   // round-to-nearest-even
    return (unsigned short)(u >> 16);
}

// f32 NCHW -> bf16 channels-last with 1-px zero halo: xp[(py*WP+px)*C + c]
__global__ void pack_input_bf16(const float* __restrict__ x,
                                unsigned short* __restrict__ xp) {
    int idx = blockIdx.x * blockDim.x + threadIdx.x;
    const int total = HP * WP * Cc;
    if (idx >= total) return;
    int c  = idx & 255;
    int pp = idx >> 8;
    int px = pp % WP, py = pp / WP;
    float v = 0.f;
    if (py >= 1 && py <= Hh && px >= 1 && px <= Ww)
        v = x[(size_t)c * HW + (py - 1) * Ww + (px - 1)];
    xp[idx] = f2bf(v);
}

// w1 OIHW [256,256,3,3] -> Bt[co][tap*256+ci] bf16 (row-major N x K)
__global__ void pack_w1_bf16(const float* __restrict__ w1,
                             unsigned short* __restrict__ Bt) {
    int t = blockIdx.x * blockDim.x + threadIdx.x;
    if (t >= Cc * KTOT) return;
    int co = t / KTOT, k = t % KTOT;
    int tap = k >> 8, ci = k & 255;
    Bt[t] = f2bf(w1[(co * Cc + ci) * 9 + tap]);
}

// init 6-channel head output scratch with conv1x1 biases (atomics accumulate on top)
__global__ void init_head_out(float* __restrict__ o,
                              const float* __restrict__ b2c, const float* __restrict__ b2k,
                              const float* __restrict__ b2o, const float* __restrict__ b2r) {
    int t = blockIdx.x * blockDim.x + threadIdx.x;
    if (t >= 6 * HW) return;
    int ch = t / HW;
    float b;
    switch (ch) {
        case 0:  b = b2c[0]; break;
        case 1:  b = b2k[0]; break;
        case 2:  b = b2o[0]; break;
        case 3:  b = b2o[1]; break;
        case 4:  b = b2r[0]; break;
        default: b = b2r[1]; break;
    }
    o[t] = b;
}

// Fused Conv3x3(256->256) + bias + ReLU + Conv1x1(256->cout) for one head.
// Block: 256 threads = 8 waves as 2(M) x 4(N): 64 pixels x 256 hidden channels.
// Wave tile: M=32 x N=64 -> 2 A frags x 4 B frags = 8 WMMA per K-step of 32
// on 12 b128 loads (1.5 loads/WMMA). Double-buffered stages, emission order
// pinned with sched_barrier so WMMAs wait only on the PREVIOUS stage's loads.
__global__ __launch_bounds__(256) void conv_head_wmma(
    const unsigned short* __restrict__ xp,   // padded bf16 input [HP][WP][C]
    const unsigned short* __restrict__ Bt,   // bf16 weights [256][2304]
    const float* __restrict__ b1,            // [256]
    const float* __restrict__ w2,            // [cout][256]
    int cout,
    float* __restrict__ out)                 // [cout][HW], pre-biased, atomic accum
{
    const int lane = threadIdx.x & 31;
    const int wv   = threadIdx.x >> 5;
    const int wrow = wv >> 2;                // 0..1
    const int wcol = wv & 3;                 // 0..3
    const int n0   = wcol * 64;
    const int l15  = lane & 15;
    const int half = lane >> 4;
    const int pwave = blockIdx.x * 64 + wrow * 32;   // first pixel of this wave's M=32

    // 2 A base pointers (one per 16-pixel subtile); halo padding keeps every
    // tap in-bounds; lane half offset folded in. Tap/kk become immediates.
    const unsigned short* abase[2];
#pragma unroll
    for (int m = 0; m < 2; ++m) {
        int pix = pwave + m * 16 + l15;
        int y = pix / Ww, x = pix % Ww;
        abase[m] = xp + ((size_t)y * WP + x) * Cc + 8 * half;
    }
    // 4 B base pointers (row-major N x K), lane half offset folded in.
    const unsigned short* bbase[4];
#pragma unroll
    for (int i = 0; i < 4; ++i)
        bbase[i] = Bt + (size_t)(n0 + 16 * i + l15) * KTOT + 8 * half;

    v8f acc[2][4] = {};
    FragU a[2][2];     // [stage][m]
    FragU b[2][4];     // [stage][i]

    auto load_stage = [&](int s, int it) {
        const int t = it >> 3, kk = it & 7;
        // A 16x32 bf16 fragment (ISA 7.12.2): u[0]=K{ci0..+7}(+8*half), u[1]=+16
        const int ao = (t / 3) * (WP * Cc) + (t % 3) * Cc + kk * 32;
#pragma unroll
        for (int m = 0; m < 2; ++m) {
            a[s][m].u[0] = *(const uint4*)(abase[m] + ao);
            a[s][m].u[1] = *(const uint4*)(abase[m] + ao + 16);
        }
        // B 32x16 bf16 fragment: lane = N, u[0]=K{k0..+7}(+8*half), u[1]=+16
#pragma unroll
        for (int i = 0; i < 4; ++i) {
            b[s][i].u[0] = *(const uint4*)(bbase[i] + it * 32);
            b[s][i].u[1] = *(const uint4*)(bbase[i] + it * 32 + 16);
        }
    };

    load_stage(0, 0);
#pragma unroll
    for (int it = 0; it < 72; ++it) {
        const int cur = it & 1, nxt = cur ^ 1;
        // Issue next-stage loads first. For it==71 this prefetches a dead
        // stage (addresses stay inside the workspace; data never consumed),
        // keeping the unrolled stream branch-free.
        load_stage(nxt, it + 1);
        __builtin_amdgcn_sched_barrier(0);   // loads must stay above the MMAs
#pragma unroll
        for (int m = 0; m < 2; ++m)
#pragma unroll
            for (int i = 0; i < 4; ++i)
                acc[m][i] = __builtin_amdgcn_wmma_f32_16x16x32_bf16(
                    false, a[cur][m].v, false, b[cur][i].v, (short)0,
                    acc[m][i], false, false);
        __builtin_amdgcn_sched_barrier(0);   // MMAs must not sink below next loads
    }

    // Epilogue: bias + ReLU, then 1x1 conv partial dot, lane-reduce, atomic accum.
    float b1v[4], w2v[2][4];
#pragma unroll
    for (int i = 0; i < 4; ++i) {
        int n = n0 + 16 * i + l15;
        b1v[i] = b1[n];
#pragma unroll
        for (int c = 0; c < 2; ++c)
            w2v[c][i] = (c < cout) ? w2[c * Cc + n] : 0.f;
    }
#pragma unroll
    for (int m = 0; m < 2; ++m) {
        const int pb = pwave + m * 16;
#pragma unroll
        for (int r = 0; r < 8; ++r) {
            float rl[4];
#pragma unroll
            for (int i = 0; i < 4; ++i) {
                float v = acc[m][i][r] + b1v[i];
                rl[i] = v > 0.f ? v : 0.f;
            }
            for (int c = 0; c < cout; ++c) {
                float s = rl[0] * w2v[c][0] + rl[1] * w2v[c][1]
                        + rl[2] * w2v[c][2] + rl[3] * w2v[c][3];
                s += __shfl_xor(s, 1, 32);
                s += __shfl_xor(s, 2, 32);
                s += __shfl_xor(s, 4, 32);
                s += __shfl_xor(s, 8, 32);   // sum over 16-lane half (16 N values)
                if (l15 == 0) {
                    int p = pb + r + 8 * half;   // C layout: VGPR r -> M=r / M=8+r
                    atomicAdd(&out[(size_t)c * HW + p], s);
                }
            }
        }
    }
}

// sigmoid/clip, 1x3 width NMS, masks, coordinate maps -> all 9 outputs
__global__ void post_kernel(const float* __restrict__ h6, float* __restrict__ out) {
    int p = blockIdx.x * blockDim.x + threadIdx.x;
    if (p >= HW) return;
    int y = p / Ww, x = p % Ww;
    auto sigc = [](float v) {
        float s = 1.f / (1.f + expf(-v));
        return fminf(fmaxf(s, 1e-4f), 1.f - 1e-4f);
    };
    float hm   = sigc(h6[0 * HW + p]);
    float kp   = sigc(h6[1 * HW + p]);
    float offx = h6[2 * HW + p], offy = h6[3 * HW + p];
    float regx = h6[4 * HW + p], regy = h6[5 * HW + p];
    float hmax = kp;
    if (x > 0)      hmax = fmaxf(hmax, sigc(h6[1 * HW + p - 1]));
    if (x < Ww - 1) hmax = fmaxf(hmax, sigc(h6[1 * HW + p + 1]));
    float nms = (hmax == kp) ? kp : 0.f;

    out[p]            = hm;                 // hm [1,1,H,W]
    out[HW + p]       = kp;                 // kpts_hm
    out[2 * HW + p]   = offx;               // pts_offset ch0
    out[3 * HW + p]   = offy;               // pts_offset ch1
    out[4 * HW + p]   = regx;               // int_offset ch0
    out[5 * HW + p]   = regy;               // int_offset ch1
    out[6 * HW + p]   = nms;                // heat_nms
    out[7 * HW + x * Hh + y] = ((offy < 1.0f) && (nms > 0.4f)) ? 1.f : 0.f; // mask_low [W,H]
    float* root = out + 8 * HW;             // root_mat [H,W,2]
    root[p * 2 + 0] = (float)x + offx;
    root[p * 2 + 1] = (float)y + offy;
    float* alin = out + 10 * HW;            // align_mat [H,W,2]
    alin[p * 2 + 0] = (float)x + regx;
    alin[p * 2 + 1] = (float)y + regy;
    float* km = out + 12 * HW;              // kpt_mask [H,W,2]
    float kmv = (nms > 0.4f) ? 1.f : 0.f;
    km[p * 2 + 0] = kmv;
    km[p * 2 + 1] = kmv;
}

extern "C" void kernel_launch(void* const* d_in, const int* in_sizes, int n_in,
                              void* d_out, int out_size, void* d_ws, size_t ws_size,
                              hipStream_t stream) {
    const float* f_hm = (const float*)d_in[0];
    const float* w1s[4] = {(const float*)d_in[1],  (const float*)d_in[5],
                           (const float*)d_in[9],  (const float*)d_in[13]};
    const float* b1s[4] = {(const float*)d_in[2],  (const float*)d_in[6],
                           (const float*)d_in[10], (const float*)d_in[14]};
    const float* w2s[4] = {(const float*)d_in[3],  (const float*)d_in[7],
                           (const float*)d_in[11], (const float*)d_in[15]};
    const float* b2s[4] = {(const float*)d_in[4],  (const float*)d_in[8],
                           (const float*)d_in[12], (const float*)d_in[16]};

    // workspace layout (~39.6 MB):
    //   [0)                      bf16 padded input  HP*WP*C        (33,343,488 B)
    //   [33,343,488)             bf16 packed weights 4*256*2304    ( 4,718,592 B)
    //   [38,062,080)             f32 head-1x1 scratch 6*HW         ( 1,536,000 B)
    char* ws = (char*)d_ws;
    unsigned short* xp = (unsigned short*)ws;
    unsigned short* Bt = (unsigned short*)(ws + (size_t)HP * WP * Cc * 2);
    float* h6 = (float*)(ws + (size_t)HP * WP * Cc * 2 + (size_t)4 * Cc * KTOT * 2);

    {
        int total = HP * WP * Cc;
        pack_input_bf16<<<(total + 255) / 256, 256, 0, stream>>>(f_hm, xp);
    }
    for (int h = 0; h < 4; ++h)
        pack_w1_bf16<<<(Cc * KTOT + 255) / 256, 256, 0, stream>>>(
            w1s[h], Bt + (size_t)h * Cc * KTOT);
    init_head_out<<<(6 * HW + 255) / 256, 256, 0, stream>>>(h6, b2s[0], b2s[1], b2s[2], b2s[3]);

    const int chbase[4] = {0, 1, 2, 4};
    const int couts[4]  = {1, 1, 2, 2};
    for (int h = 0; h < 4; ++h)
        conv_head_wmma<<<HW / 64, 256, 0, stream>>>(
            xp, Bt + (size_t)h * Cc * KTOT, b1s[h], w2s[h], couts[h],
            h6 + (size_t)chbase[h] * HW);

    post_kernel<<<(HW + 255) / 256, 256, 0, stream>>>(h6, (float*)d_out);
}
